// ReHDM_90177133347076
// MI455X (gfx1250) — matrix operand order repaired and checked by reading, not compiled, not run on hardware
//
#include <hip/hip_runtime.h>
#include <math.h>

// ---------------- types ----------------
typedef _Float16 h16;
typedef __attribute__((ext_vector_type(8)))  _Float16 v8h;
typedef __attribute__((ext_vector_type(16))) _Float16 v16h;
typedef __attribute__((ext_vector_type(8)))  float    v8f;

#define BT    65536      // 512*128 tokens
#define BB    512
#define TT    128
#define DD    192
#define HH    4
#define DHH   48
#define DFF_  256
#define NREG_ 100
#define SCALE_DH 0.14433756729740643f  // 1/sqrt(48)

__device__ __forceinline__ v8f zero_v8f() {
  v8f z;
#pragma unroll
  for (int i = 0; i < 8; ++i) z[i] = 0.f;
  return z;
}
__device__ __forceinline__ v8h zero_v8h() {
  v8h z;
#pragma unroll
  for (int i = 0; i < 8; ++i) z[i] = (_Float16)0.f;
  return z;
}
__device__ __forceinline__ v16h cat8(v8h lo, v8h hi) {
  v16h r;
#pragma unroll
  for (int i = 0; i < 8; ++i) { r[i] = lo[i]; r[i + 8] = hi[i]; }
  return r;
}
__device__ __forceinline__ float wred_sum(float v) {
#pragma unroll
  for (int o = 16; o > 0; o >>= 1) v += __shfl_xor(v, o, 32);
  return v;
}
__device__ __forceinline__ float wred_max(float v) {
#pragma unroll
  for (int o = 16; o > 0; o >>= 1) v = fmaxf(v, __shfl_xor(v, o, 32));
  return v;
}
// ---- CDNA5 async global->LDS copy (ASYNCcnt-tracked), per ISA 08_async_tensor ----
__device__ __forceinline__ uint32_t lds_off32(const void* p) {
  return (uint32_t)(uintptr_t)p;   // low 32 bits of generic ptr to LDS == LDS byte offset
}
__device__ __forceinline__ void async_b128(uint32_t lds_addr, const void* gaddr) {
  asm volatile("global_load_async_to_lds_b128 %0, %1, off"
               :: "v"(lds_addr), "v"(gaddr) : "memory");
}
__device__ __forceinline__ void wait_async0() {
  asm volatile("s_wait_asynccnt 0" ::: "memory");
}

// ---------------- embedding gather ----------------
__global__ void k_embed(const int* __restrict__ i0, const int* __restrict__ i1,
                        const int* __restrict__ i2, const int* __restrict__ i3,
                        const int* __restrict__ i4, const int* __restrict__ i5,
                        const float* __restrict__ e0, const float* __restrict__ e1,
                        const float* __restrict__ e2, const float* __restrict__ e3,
                        const float* __restrict__ e4, const float* __restrict__ e5,
                        float* __restrict__ Xf, h16* __restrict__ Xh, int rows) {
  int gid = blockIdx.x * 256 + threadIdx.x;
  if (gid >= rows * DD) return;
  int row = gid / DD, c = gid - row * DD;
  int s = c >> 5, j = c & 31;
  const int* ip[6]   = { i0, i1, i2, i3, i4, i5 };
  const float* ep[6] = { e0, e1, e2, e3, e4, e5 };
  float v = ep[s][(size_t)ip[s][row] * 32 + j];
  Xf[gid] = v;
  Xh[gid] = (h16)v;
}

// ---------------- generic f16 WMMA GEMM:  C = A[M,K] * W[N,K]^T + bias ----------------
// Double-buffered async-to-LDS pipeline.
#define GBM 128
#define GBN 64
#define GBK 64
#define LSA 72
#define LSB 72
__global__ __launch_bounds__(256) void k_gemm(const h16* __restrict__ A,
                                              const h16* __restrict__ W,
                                              const float* __restrict__ bias,
                                              float* __restrict__ Cf, h16* __restrict__ Ch,
                                              int M, int N, int K, int relu) {
  __shared__ alignas(16) h16 sA[2][GBM * LSA];
  __shared__ alignas(16) h16 sB[2][GBN * LSB];
  int tid = threadIdx.x;
  int m0 = blockIdx.x * GBM, n0 = blockIdx.y * GBN;
  int lane = tid & 31, w = tid >> 5;
  int wm = w & 3, wn = w >> 2;
  int l15 = lane & 15, lhi = lane >> 4;

  auto issue_stage = [&](int k0, int buf) {
#pragma unroll
    for (int it = 0; it < 4; ++it) {                        // A tile 128x64 halves
      int idx = tid + it * 256;
      int r = idx >> 3, c8 = (idx & 7) * 8;
      async_b128(lds_off32(&sA[buf][r * LSA + c8]),
                 A + (size_t)(m0 + r) * K + k0 + c8);
    }
#pragma unroll
    for (int it = 0; it < 2; ++it) {                        // W tile 64x64 halves
      int idx = tid + it * 256;
      int r = idx >> 3, c8 = (idx & 7) * 8;
      if (n0 + r < N) {
        async_b128(lds_off32(&sB[buf][r * LSB + c8]),
                   W + (size_t)(n0 + r) * K + k0 + c8);
      } else {
        *(v8h*)(&sB[buf][r * LSB + c8]) = zero_v8h();       // ragged N (classifier)
      }
    }
  };

  v8f c[2][2];
#pragma unroll
  for (int i = 0; i < 2; ++i)
#pragma unroll
    for (int j = 0; j < 2; ++j) c[i][j] = zero_v8f();

  int S = K / GBK;
  issue_stage(0, 0);
  wait_async0();
  __syncthreads();
  for (int s = 0; s < S; ++s) {
    if (s + 1 < S) issue_stage((s + 1) * GBK, (s + 1) & 1); // prefetch next stage
    const h16* cA = &sA[s & 1][0];
    const h16* cB = &sB[s & 1][0];
#pragma unroll
    for (int kk = 0; kk < GBK; kk += 32) {
      v16h a[2], b[2];
      int ak0 = kk + (lhi << 3);
#pragma unroll
      for (int i = 0; i < 2; ++i) {
        int row = wm * 32 + i * 16 + l15;
        a[i] = cat8(*(const v8h*)(cA + row * LSA + ak0),
                    *(const v8h*)(cA + row * LSA + ak0 + 16));
      }
      int bk0 = kk + (lhi << 3);
#pragma unroll
      for (int j = 0; j < 2; ++j) {
        int col = wn * 32 + j * 16 + l15;
        b[j] = cat8(*(const v8h*)(cB + col * LSB + bk0),
                    *(const v8h*)(cB + col * LSB + bk0 + 16));
      }
#pragma unroll
      for (int i = 0; i < 2; ++i)
#pragma unroll
        for (int j = 0; j < 2; ++j)
          c[i][j] = __builtin_amdgcn_wmma_f32_16x16x32_f16(
              false, a[i], false, b[j], (short)0, c[i][j], false, false);
    }
    wait_async0();
    __syncthreads();
  }
#pragma unroll
  for (int j = 0; j < 2; ++j) {
    int col = n0 + wn * 32 + j * 16 + l15;
    if (col >= N) continue;
    float bb = bias ? bias[col] : 0.f;
#pragma unroll
    for (int i = 0; i < 2; ++i) {
#pragma unroll
      for (int r = 0; r < 8; ++r) {
        int row = m0 + wm * 32 + i * 16 + r + (lhi << 3);
        float v = c[i][j][r] + bb;
        if (relu) v = fmaxf(v, 0.f);
        if (Cf) Cf[(size_t)row * N + col] = v;
        if (Ch) Ch[(size_t)row * N + col] = (h16)v;
      }
    }
  }
}

// ---------------- POI multi-head attention (per b,h block), WMMA ----------------
__global__ __launch_bounds__(128) void k_attn(const h16* __restrict__ qkv,
                                              const int* __restrict__ mask,
                                              h16* __restrict__ O) {
  constexpr int LQ = 72, LV = 136;
  __shared__ alignas(16) h16 sQ[128 * LQ];
  __shared__ alignas(16) h16 sK[128 * LQ];
  __shared__ alignas(16) h16 sVt[64 * LV];
  __shared__ alignas(16) float sS[4 * 32 * 128];
  int tid = threadIdx.x;
  int b = blockIdx.x >> 2, hd = blockIdx.x & 3;
  {                                        // stage q/k row-major (async), v transposed
    int t = tid;
    const h16* qb = qkv + ((size_t)(b * TT + t)) * 576 + hd * DHH;
#pragma unroll
    for (int cc = 0; cc < 48; cc += 8) {
      async_b128(lds_off32(sQ + t * LQ + cc), qb + cc);
      async_b128(lds_off32(sK + t * LQ + cc), qb + 192 + cc);
    }
    v8h z = zero_v8h();
    *(v8h*)(sQ + t * LQ + 48) = z; *(v8h*)(sQ + t * LQ + 56) = z;
    *(v8h*)(sK + t * LQ + 48) = z; *(v8h*)(sK + t * LQ + 56) = z;
#pragma unroll
    for (int cc = 0; cc < 48; ++cc) sVt[cc * LV + t] = qb[384 + cc];
#pragma unroll
    for (int cc = 48; cc < 64; ++cc) sVt[cc * LV + t] = (h16)0.f;
  }
  wait_async0();
  __syncthreads();
  int wv = tid >> 5, lane = tid & 31;
  int l15 = lane & 15, lhi = lane >> 4;
  int qb0 = wv * 32;
  float* myS = sS + wv * 32 * 128;

  v16h aq[2][2];
#pragma unroll
  for (int kk2 = 0; kk2 < 2; ++kk2) {
    int ak0 = kk2 * 32 + (lhi << 3);
#pragma unroll
    for (int i = 0; i < 2; ++i) {
      int row = qb0 + i * 16 + l15;
      aq[kk2][i] = cat8(*(const v8h*)(sQ + row * LQ + ak0),
                        *(const v8h*)(sQ + row * LQ + ak0 + 16));
    }
  }
#pragma unroll
  for (int tj = 0; tj < 8; ++tj) {                          // S = Q K^T
    v8f c0 = zero_v8f(), c1 = zero_v8f();
#pragma unroll
    for (int kk2 = 0; kk2 < 2; ++kk2) {
      int bk0 = kk2 * 32 + (lhi << 3);
      int colr = tj * 16 + l15;
      v16h bk = cat8(*(const v8h*)(sK + colr * LQ + bk0),
                     *(const v8h*)(sK + colr * LQ + bk0 + 16));
      c0 = __builtin_amdgcn_wmma_f32_16x16x32_f16(false, aq[kk2][0], false, bk, (short)0, c0, false, false);
      c1 = __builtin_amdgcn_wmma_f32_16x16x32_f16(false, aq[kk2][1], false, bk, (short)0, c1, false, false);
    }
#pragma unroll
    for (int r = 0; r < 8; ++r) {
      int cc = tj * 16 + l15;
      int r0 = r + (lhi << 3);
      myS[r0 * 128 + cc]        = c0[r];
      myS[(16 + r0) * 128 + cc] = c1[r];
    }
  }
  __syncthreads();
  {                                                          // softmax, lane==row
    const int* mk = mask + b * TT;
    int row = lane;
    float m = -3.0e38f;
    for (int j = 0; j < 128; ++j) {
      float s = mk[j] ? myS[row * 128 + j] * SCALE_DH : -1.0e9f;
      m = fmaxf(m, s);
    }
    float sum = 0.f;
    for (int j = 0; j < 128; ++j) {
      float s = mk[j] ? myS[row * 128 + j] * SCALE_DH : -1.0e9f;
      float e = __expf(s - m);
      myS[row * 128 + j] = e;
      sum += e;
    }
    float rs = 1.f / sum;
    h16* Pr = (h16*)myS;                                     // in-place f32 -> f16
    for (int j = 0; j < 128; ++j)
      Pr[row * 256 + j] = (h16)(myS[row * 128 + j] * rs);
  }
  __syncthreads();
  h16* Pw = (h16*)myS;                                       // O = P V
#pragma unroll
  for (int tj = 0; tj < 3; ++tj) {
    v8f c0 = zero_v8f(), c1 = zero_v8f();
#pragma unroll
    for (int kk = 0; kk < 128; kk += 32) {
      int ak0 = kk + (lhi << 3);
      v16h a0 = cat8(*(const v8h*)(Pw + l15 * 256 + ak0),
                     *(const v8h*)(Pw + l15 * 256 + ak0 + 16));
      v16h a1 = cat8(*(const v8h*)(Pw + (16 + l15) * 256 + ak0),
                     *(const v8h*)(Pw + (16 + l15) * 256 + ak0 + 16));
      int bk0 = kk + (lhi << 3);
      int coldh = tj * 16 + l15;
      v16h bvv = cat8(*(const v8h*)(sVt + coldh * LV + bk0),
                      *(const v8h*)(sVt + coldh * LV + bk0 + 16));
      c0 = __builtin_amdgcn_wmma_f32_16x16x32_f16(false, a0, false, bvv, (short)0, c0, false, false);
      c1 = __builtin_amdgcn_wmma_f32_16x16x32_f16(false, a1, false, bvv, (short)0, c1, false, false);
    }
#pragma unroll
    for (int r = 0; r < 8; ++r) {
      int q0 = qb0 + r + (lhi << 3);
      int dh = tj * 16 + l15;
      O[((size_t)(b * TT + q0)) * DD + hd * DHH + dh]        = (h16)c0[r];
      O[((size_t)(b * TT + q0 + 16)) * DD + hd * DHH + dh]   = (h16)c1[r];
    }
  }
}

// ---------------- LayerNorm (row length 192), wave per row ----------------
// rmode: 0 none, 1 full residual, 2 broadcast-vector residual
__global__ __launch_bounds__(256) void k_ln(const float* __restrict__ X,
                                            const float* __restrict__ R, int rmode,
                                            int relu_pre,
                                            const float* __restrict__ g,
                                            const float* __restrict__ bt,
                                            float* __restrict__ Yf, h16* __restrict__ Yh,
                                            int rows) {
  int w = threadIdx.x >> 5, lane = threadIdx.x & 31;
  int row = blockIdx.x * 8 + w;
  if (row >= rows) return;
  const float* x = X + (size_t)row * DD;
  float v[6];
#pragma unroll
  for (int i = 0; i < 6; ++i) {
    int c = lane + i * 32;
    float t = x[c];
    if (rmode == 1) t += R[(size_t)row * DD + c];
    else if (rmode == 2) t += R[c];
    if (relu_pre) t = fmaxf(t, 0.f);
    v[i] = t;
  }
  float s = 0.f;
#pragma unroll
  for (int i = 0; i < 6; ++i) s += v[i];
  s = wred_sum(s);
  float mean = s * (1.f / 192.f);
  float q = 0.f;
#pragma unroll
  for (int i = 0; i < 6; ++i) { float d = v[i] - mean; q += d * d; }
  q = wred_sum(q);
  float inv = rsqrtf(q * (1.f / 192.f) + 1e-5f);
#pragma unroll
  for (int i = 0; i < 6; ++i) {
    int c = lane + i * 32;
    float y = (v[i] - mean) * inv * g[c] + bt[c];
    if (Yf) Yf[(size_t)row * DD + c] = y;
    if (Yh) Yh[(size_t)row * DD + c] = (h16)y;
  }
}

// ---------------- misc elementwise ----------------
__global__ void k_f2h(const float* __restrict__ s, h16* __restrict__ d, int n) {
  int i = blockIdx.x * 256 + threadIdx.x;
  if (i < n) d[i] = (h16)s[i];
}
__global__ void k_copyf(const float* __restrict__ s, float* __restrict__ d, int n) {
  int i = blockIdx.x * 256 + threadIdx.x;
  if (i < n) d[i] = s[i];
}
__global__ void k_addvec_h(const float* __restrict__ X, const float* __restrict__ V,
                           h16* __restrict__ Y, int n) {
  int i = blockIdx.x * 256 + threadIdx.x;
  if (i < n) Y[i] = (h16)(X[i] + V[i % DD]);
}
__global__ void k_mix_h(const float* __restrict__ Xa, const float* __restrict__ Xg,
                        h16* __restrict__ Z, int n) {
  int i = blockIdx.x * 256 + threadIdx.x;
  if (i < n) Z[i] = (h16)(0.5f * Xa[i] + 0.5f * Xg[i]);
}
// tiny dense: C[r,n] = A[r,:K] . W[n,:K] + bias[n];  grid=R blocks, block=N threads
__global__ void k_tinylin(const float* __restrict__ A, const float* __restrict__ W,
                          const float* __restrict__ bias, float* __restrict__ C,
                          int K, int N) {
  int r = blockIdx.x, n = threadIdx.x;
  const float* a = A + (size_t)r * K;
  const float* wr = W + (size_t)n * K;
  float acc = bias ? bias[n] : 0.f;
  for (int k = 0; k < K; ++k) acc += a[k] * wr[k];
  C[(size_t)r * N + n] = acc;
}

// ---------------- init_traj attention pooling (block per b) ----------------
__global__ __launch_bounds__(128) void k_pool(const float* __restrict__ qv,
                                              const h16* __restrict__ kv,  // [BT,384] k|v
                                              const int* __restrict__ mask,
                                              h16* __restrict__ Oh) {
  int b = blockIdx.x, tid = threadIdx.x;
  __shared__ float sq[DD];
  __shared__ float sa[HH][TT];
  for (int c = tid; c < DD; c += 128) sq[c] = qv[c];
  __syncthreads();
  {
    int t = tid;
    const h16* kr = kv + ((size_t)(b * TT + t)) * 384;
    int mk = mask[b * TT + t];
#pragma unroll
    for (int hh = 0; hh < HH; ++hh) {
      float s = 0.f;
      for (int d = 0; d < DHH; ++d) s += sq[hh * DHH + d] * (float)kr[hh * DHH + d];
      sa[hh][t] = mk ? s * SCALE_DH : -1.0e9f;
    }
  }
  __syncthreads();
  int wv = tid >> 5, lane = tid & 31;
  if (wv < HH) {
    float m = -3.0e38f;
    for (int j = lane; j < TT; j += 32) m = fmaxf(m, sa[wv][j]);
    m = wred_max(m);
    float sum = 0.f;
    for (int j = lane; j < TT; j += 32) { float e = __expf(sa[wv][j] - m); sa[wv][j] = e; sum += e; }
    sum = wred_sum(sum);
    float rs = 1.f / sum;
    for (int j = lane; j < TT; j += 32)
      sa[wv][j] = mask[b * TT + j] ? sa[wv][j] * rs : 0.f;
  }
  __syncthreads();
  for (int d = tid; d < DD; d += 128) {
    int hh = d / DHH;
    float acc = 0.f;
    for (int t2 = 0; t2 < TT; ++t2)
      acc += sa[hh][t2] * (float)kv[((size_t)(b * TT + t2)) * 384 + 192 + d];
    Oh[(size_t)b * DD + d] = (h16)acc;
  }
}

// ---------------- e2e cross attention, algebraically decomposed (block per b) ----------------
__global__ __launch_bounds__(256) void k_e2e_attn(const float* __restrict__ q,
                                                  const float* __restrict__ kn,
                                                  const float* __restrict__ vn,
                                                  const float* __restrict__ etk,
                                                  const float* __restrict__ etv,
                                                  const int* __restrict__ adj,
                                                  const int* __restrict__ et,
                                                  float* __restrict__ Of, h16* __restrict__ Oh) {
  int b = blockIdx.x, tid = threadIdx.x;
  __shared__ float sq[DD];
  __shared__ float qe[HH][2];
  __shared__ float sa[HH][BB];
  __shared__ float saet[HH][2];
  for (int c = tid; c < DD; c += 256) sq[c] = q[(size_t)b * DD + c];
  __syncthreads();
  if (tid < 8) {
    int hh = tid >> 1, e = tid & 1;
    float s = 0.f;
    for (int d = 0; d < DHH; ++d) s += sq[hh * DHH + d] * etk[e * DD + hh * DHH + d];
    qe[hh][e] = s;
  }
  __syncthreads();
  for (int n = tid; n < BB; n += 256) {
    int a_ = adj[(size_t)b * BB + n];
    int ee = et[(size_t)b * BB + n];
    const float* kr = kn + (size_t)n * DD;
#pragma unroll
    for (int hh = 0; hh < HH; ++hh) {
      float s = 0.f;
      for (int d = 0; d < DHH; ++d) s += sq[hh * DHH + d] * kr[hh * DHH + d];
      s = (s + qe[hh][ee]) * SCALE_DH;
      sa[hh][n] = a_ ? s : -1.0e9f;
    }
  }
  __syncthreads();
  int wv = tid >> 5, lane = tid & 31;
  if (wv < HH) {
    float m = -3.0e38f;
    for (int j = lane; j < BB; j += 32) m = fmaxf(m, sa[wv][j]);
    m = wred_max(m);
    float sum = 0.f;
    for (int j = lane; j < BB; j += 32) { float e = __expf(sa[wv][j] - m); sa[wv][j] = e; sum += e; }
    sum = wred_sum(sum);
    float rs = 1.f / sum;
    float a0 = 0.f, a1 = 0.f;
    for (int j = lane; j < BB; j += 32) {
      int ad = adj[(size_t)b * BB + j];
      float p = ad ? sa[wv][j] * rs : 0.f;
      sa[wv][j] = p;
      if (ad) { if (et[(size_t)b * BB + j]) a1 += p; else a0 += p; }
    }
    a0 = wred_sum(a0); a1 = wred_sum(a1);
    if (lane == 0) { saet[wv][0] = a0; saet[wv][1] = a1; }
  }
  __syncthreads();
  for (int d = tid; d < DD; d += 256) {
    int hh = d / DHH;
    float acc = saet[hh][0] * etv[d] + saet[hh][1] * etv[DD + d];
    for (int n = 0; n < BB; ++n) acc += sa[hh][n] * vn[(size_t)n * DD + d];
    if (Of) Of[(size_t)b * DD + d] = acc;
    Oh[(size_t)b * DD + d] = (h16)acc;
  }
}

// ---------------- host ----------------
extern "C" void kernel_launch(void* const* d_in, const int* in_sizes, int n_in,
                              void* d_out, int out_size, void* d_ws, size_t ws_size,
                              hipStream_t stream) {
  // input order (setup_inputs insertion order)
  enum { I_USER = 0, I_POI, I_CAT, I_HOUR, I_DAY, I_QK, I_TMASK,
         I_CUSER, I_CPOI, I_CCAT, I_CHOUR, I_CDAY, I_CQK, I_CMASK,
         I_ADJ, I_ET,
         P_EU, P_EP, P_EC, P_EH, P_ED, P_EQ, P_THETA,
         P_POI_WQ, P_POI_BQ, P_POI_WK, P_POI_BK, P_POI_WV, P_POI_BV, P_POI_WO, P_POI_BO,
         P_V2E_WQ, P_V2E_BQ, P_V2E_WK, P_V2E_BK, P_V2E_WV, P_V2E_BV, P_V2E_WO, P_V2E_BO,
         P_E2E_WQ, P_E2E_BQ, P_E2E_WK, P_E2E_BK, P_E2E_WV, P_E2E_BV, P_E2E_WO, P_E2E_BO,
         P_LN1G, P_LN1B, P_LN2G, P_LN2B,
         P_FFW1, P_FFB1, P_FFW2, P_FFB2,
         P_V2E_ET, P_E2E_ET,
         P_V2E_M1W, P_V2E_M1B, P_V2E_M2W, P_V2E_M2B,
         P_V2E_LNG, P_V2E_LNB, P_V2E_PG, P_V2E_PB,
         P_E2E_M1W, P_E2E_M1B, P_E2E_M2W, P_E2E_M2B,
         P_ALW, P_ALB, P_E2E_PG, P_E2E_PB, P_CLSW, P_CLSB };
  auto F = [&](int i) { return (const float*)d_in[i]; };
  auto I = [&](int i) { return (const int*)d_in[i]; };

  char* wsp = (char*)d_ws;
  size_t off = 0;
  auto alloc = [&](size_t bytes) -> void* {
    void* p = wsp + off;
    off += (bytes + 255) & ~(size_t)255;
    return p;
  };
  // big (shared between target/collab passes)
  float* x32   = (float*)alloc((size_t)BT * DD * 4);
  h16*   x16   = (h16*)  alloc((size_t)BT * DD * 2);
  h16*   qkv16 = (h16*)  alloc((size_t)BT * 576 * 2);
  h16*   o16   = (h16*)  alloc((size_t)BT * DD * 2);
  float* t32a  = (float*)alloc((size_t)BT * DD * 4);   // also reused as kv16 [BT,384] h16
  float* x1_32 = (float*)alloc((size_t)BT * DD * 4);
  h16*   x1_16 = (h16*)  alloc((size_t)BT * DD * 2);
  h16*   ff1   = (h16*)  alloc((size_t)BT * DFF_ * 2);
  // f16 weights
  h16* Wqkv  = (h16*)alloc(576 * DD * 2);
  h16* Wpo   = (h16*)alloc(DD * DD * 2);
  h16* Wff1  = (h16*)alloc(DFF_ * DD * 2);
  h16* Wff2  = (h16*)alloc(DD * DFF_ * 2);
  h16* Wkv2  = (h16*)alloc(384 * DD * 2);
  h16* Wvo   = (h16*)alloc(DD * DD * 2);
  h16* Wvm1  = (h16*)alloc(DD * DD * 2);
  h16* Wvm2  = (h16*)alloc(DD * DD * 2);
  h16* Weq   = (h16*)alloc(DD * DD * 2);
  h16* Wek   = (h16*)alloc(DD * DD * 2);
  h16* Wev   = (h16*)alloc(DD * DD * 2);
  h16* Weo   = (h16*)alloc(DD * DD * 2);
  h16* Wal   = (h16*)alloc(DD * DD * 2);
  h16* Wem1  = (h16*)alloc(DD * DD * 2);
  h16* Wem2  = (h16*)alloc(DD * DD * 2);
  h16* Wcls  = (h16*)alloc(NREG_ * DD * 2);
  float* bqkv = (float*)alloc(576 * 4);
  float* bkv2 = (float*)alloc(384 * 4);
  // small f32 / f16
  float* qtheta = (float*)alloc(DD * 4);
  h16*   pool16 = (h16*)  alloc((size_t)BB * DD * 2);
  float* S1 = (float*)alloc((size_t)BB * DD * 4);
  float* S2 = (float*)alloc((size_t)BB * DD * 4);
  h16*   S3h = (h16*)alloc((size_t)BB * DD * 2);
  h16*   S4h = (h16*)alloc((size_t)BB * DD * 2);
  float* S5 = (float*)alloc((size_t)BB * DD * 4);
  float* hT32 = (float*)alloc((size_t)BB * DD * 4);
  h16*   hT16 = (h16*)  alloc((size_t)BB * DD * 2);
  float* hN32 = (float*)alloc((size_t)BB * DD * 4);
  h16*   hN16 = (h16*)  alloc((size_t)BB * DD * 2);
  float* E1 = (float*)alloc((size_t)BB * DD * 4);   // q
  float* E2 = (float*)alloc((size_t)BB * DD * 4);   // kn
  float* E3 = (float*)alloc((size_t)BB * DD * 4);   // vn
  float* E4 = (float*)alloc((size_t)BB * DD * 4);   // attn out f32
  h16*   E4h = (h16*)alloc((size_t)BB * DD * 2);
  float* E5 = (float*)alloc((size_t)BB * DD * 4);   // g
  float* E6 = (float*)alloc((size_t)BB * DD * 4);   // align
  h16*   E7h = (h16*)alloc((size_t)BB * DD * 2);    // gated
  h16*   E8h = (h16*)alloc((size_t)BB * DD * 2);    // mlp m1
  float* E9 = (float*)alloc((size_t)BB * DD * 4);   // mlp m2
  float* E10 = (float*)alloc((size_t)BB * DD * 4);
  h16*   E10h = (h16*)alloc((size_t)BB * DD * 2);
  float* ETK = (float*)alloc(2 * DD * 4);
  float* ETV = (float*)alloc(2 * DD * 4);
  (void)ws_size; (void)n_in; (void)in_sizes; (void)out_size;

  auto cast = [&](const float* s, h16* d, int n) {
    k_f2h<<<(n + 255) / 256, 256, 0, stream>>>(s, d, n);
  };
  auto copyf = [&](const float* s, float* d, int n) {
    k_copyf<<<(n + 255) / 256, 256, 0, stream>>>(s, d, n);
  };
  auto gemm = [&](const h16* A, const h16* Wt, const float* bias, float* Cf, h16* Ch,
                  int M, int N, int K, int relu) {
    dim3 g((M + GBM - 1) / GBM, (N + GBN - 1) / GBN);
    k_gemm<<<g, 256, 0, stream>>>(A, Wt, bias, Cf, Ch, M, N, K, relu);
  };

  // ---- convert weights to f16 (every call; deterministic) ----
  cast(F(P_POI_WQ), Wqkv + 0 * DD * DD, DD * DD);
  cast(F(P_POI_WK), Wqkv + 1 * DD * DD, DD * DD);
  cast(F(P_POI_WV), Wqkv + 2 * DD * DD, DD * DD);
  copyf(F(P_POI_BQ), bqkv + 0, DD);
  copyf(F(P_POI_BK), bqkv + DD, DD);
  copyf(F(P_POI_BV), bqkv + 2 * DD, DD);
  cast(F(P_POI_WO), Wpo, DD * DD);
  cast(F(P_FFW1), Wff1, DFF_ * DD);
  cast(F(P_FFW2), Wff2, DD * DFF_);
  cast(F(P_V2E_WK), Wkv2 + 0, DD * DD);
  cast(F(P_V2E_WV), Wkv2 + DD * DD, DD * DD);
  copyf(F(P_V2E_BK), bkv2 + 0, DD);
  copyf(F(P_V2E_BV), bkv2 + DD, DD);
  cast(F(P_V2E_WO), Wvo, DD * DD);
  cast(F(P_V2E_M1W), Wvm1, DD * DD);
  cast(F(P_V2E_M2W), Wvm2, DD * DD);
  cast(F(P_E2E_WQ), Weq, DD * DD);
  cast(F(P_E2E_WK), Wek, DD * DD);
  cast(F(P_E2E_WV), Wev, DD * DD);
  cast(F(P_E2E_WO), Weo, DD * DD);
  cast(F(P_ALW), Wal, DD * DD);
  cast(F(P_E2E_M1W), Wem1, DD * DD);
  cast(F(P_E2E_M2W), Wem2, DD * DD);
  cast(F(P_CLSW), Wcls, NREG_ * DD);

  // shared theta query for init_traj
  k_tinylin<<<1, DD, 0, stream>>>(F(P_THETA), F(P_V2E_WQ), F(P_V2E_BQ), qtheta, DD, DD);

  // ---- per-trajectory pipeline (0 = target, 1 = collab) ----
  for (int dom = 0; dom < 2; ++dom) {
    const int* idx0 = I(dom ? I_CUSER : I_USER);
    const int* idx1 = I(dom ? I_CPOI  : I_POI);
    const int* idx2 = I(dom ? I_CCAT  : I_CAT);
    const int* idx3 = I(dom ? I_CHOUR : I_HOUR);
    const int* idx4 = I(dom ? I_CDAY  : I_DAY);
    const int* idx5 = I(dom ? I_CQK   : I_QK);
    const int* msk  = I(dom ? I_CMASK : I_TMASK);
    float* hOut32 = dom ? hN32 : hT32;
    h16*   hOut16 = dom ? hN16 : hT16;

    // embed
    k_embed<<<((size_t)BT * DD + 255) / 256, 256, 0, stream>>>(
        idx0, idx1, idx2, idx3, idx4, idx5,
        F(P_EU), F(P_EP), F(P_EC), F(P_EH), F(P_ED), F(P_EQ), x32, x16, BT);
    // POI block
    gemm(x16, Wqkv, bqkv, nullptr, qkv16, BT, 576, DD, 0);
    k_attn<<<BB * HH, 128, 0, stream>>>(qkv16, msk, o16);
    gemm(o16, Wpo, F(P_POI_BO), t32a, nullptr, BT, DD, DD, 0);
    k_ln<<<BT / 8, 256, 0, stream>>>(t32a, x32, 1, 0, F(P_LN1G), F(P_LN1B), x1_32, x1_16, BT);
    gemm(x1_16, Wff1, F(P_FFB1), nullptr, ff1, BT, DFF_, DD, 1);
    gemm(ff1, Wff2, F(P_FFB2), t32a, nullptr, BT, DD, DFF_, 0);
    k_ln<<<BT / 8, 256, 0, stream>>>(t32a, x1_32, 1, 0, F(P_LN2G), F(P_LN2B), x1_32, x1_16, BT);
    // init_traj:  msg = t_rep + v2e_et[0]
    k_addvec_h<<<((size_t)BT * DD + 255) / 256, 256, 0, stream>>>(x1_32, F(P_V2E_ET), x16, BT * DD);
    gemm(x16, Wkv2, bkv2, nullptr, (h16*)t32a, BT, 384, DD, 0);      // kv packed f16
    k_pool<<<BB, 128, 0, stream>>>(qtheta, (const h16*)t32a, msk, pool16);
    gemm(pool16, Wvo, F(P_V2E_BO), S1, nullptr, BB, DD, DD, 0);
    k_ln<<<BB / 8, 256, 0, stream>>>(S1, F(P_THETA), 2, 0, F(P_V2E_LNG), F(P_V2E_LNB), S2, S3h, BB);
    gemm(S3h, Wvm1, F(P_V2E_M1B), nullptr, S4h, BB, DD, DD, 1);
    gemm(S4h, Wvm2, F(P_V2E_M2B), S5, nullptr, BB, DD, DD, 0);
    k_ln<<<BB / 8, 256, 0, stream>>>(S5, nullptr, 0, 1, F(P_V2E_PG), F(P_V2E_PB), hOut32, hOut16, BB);
  }

  // ---- e2e layer (decomposed) ----
  gemm(hT16, Weq, F(P_E2E_BQ), E1, nullptr, BB, DD, DD, 0);   // q
  gemm(hN16, Wek, F(P_E2E_BK), E2, nullptr, BB, DD, DD, 0);   // kn (+bk)
  gemm(hN16, Wev, F(P_E2E_BV), E3, nullptr, BB, DD, DD, 0);   // vn (+bv)
  k_tinylin<<<2, DD, 0, stream>>>(F(P_E2E_ET), F(P_E2E_WK), nullptr, ETK, DD, DD);
  k_tinylin<<<2, DD, 0, stream>>>(F(P_E2E_ET), F(P_E2E_WV), nullptr, ETV, DD, DD);
  k_e2e_attn<<<BB, 256, 0, stream>>>(E1, E2, E3, ETK, ETV, I(I_ADJ), I(I_ET), E4, E4h);
  gemm(E4h, Weo, F(P_E2E_BO), E5, nullptr, BB, DD, DD, 0);    // g
  gemm(hT16, Wal, F(P_ALB), E6, nullptr, BB, DD, DD, 0);      // align
  k_mix_h<<<((size_t)BB * DD + 255) / 256, 256, 0, stream>>>(E6, E5, E7h, BB * DD);
  gemm(E7h, Wem1, F(P_E2E_M1B), nullptr, E8h, BB, DD, DD, 1);
  gemm(E8h, Wem2, F(P_E2E_M2B), E9, nullptr, BB, DD, DD, 0);
  k_ln<<<BB / 8, 256, 0, stream>>>(E9, nullptr, 0, 1, F(P_E2E_PG), F(P_E2E_PB), E10, E10h, BB);
  // classifier -> d_out [512,100] f32
  gemm(E10h, Wcls, F(P_CLSB), (float*)d_out, nullptr, BB, NREG_, DD, 0);
}
